// FastPitch_79465484910889
// MI455X (gfx1250) — matrix operand order repaired and checked by reading, not compiled, not run on hardware
//
#include <hip/hip_runtime.h>

// ---------------- model constants ----------------
#define B_     32
#define S_     256
#define D_     384
#define NL_    6
#define DH_    64
#define A3_    192      // 3*N_HEAD*D_HEAD
#define DI_    1536
#define KW_    3
#define DPF_   256
#define NMEL_  80
#define TDM_   400      // MEL_MAX
#define PACE_  0.0265f
#define MAXDUR_ 75.0f
#define ATTNSC_ 0.125f  // 1/sqrt(64)

typedef unsigned int u32;
typedef __attribute__((ext_vector_type(4)))  u32    u32x4;
typedef __bf16 bf16_t;
typedef __attribute__((ext_vector_type(16))) bf16_t v16bf;
typedef __attribute__((ext_vector_type(8)))  float  v8f;

union FragU { v16bf v; u32x4 q[2]; };

__device__ __forceinline__ unsigned short f2bf(float f) {
  union { float f; u32 u; } x; x.f = f;
  u32 u = x.u;
  u32 r = (u + 0x7FFFu + ((u >> 16) & 1u)) >> 16;  // round-to-nearest-even
  return (unsigned short)r;
}

__device__ __forceinline__ float pos_emb(int t, int d, int D) {
  int half = D >> 1;
  int j = (d < half) ? d : d - half;
  float inv = expf(-(2.0f * (float)j / (float)D) * 9.210340371976184f); // ln(10000)
  float s = (float)t * inv;
  return (d < half) ? sinf(s) : cosf(s);
}

// ---------------- generic WMMA GEMM ----------------
// C[m,n] = act( alpha * sum_k A[m,k]*B[k,n] + bias[n] )
// convMode: A is im2col of X (K taps=3, pad=1) with tokens-per-batch Tconv and
//           inner dim Kin; B weight layout [N][Kin][3] (out,in,tap).
// transB:   B[k][n] = Bw[n*ldb + k]  (used for Q*K^T)
__global__ __launch_bounds__(256) void wmma_gemm(
    const float* __restrict__ A, const float* __restrict__ Bw,
    const float* __restrict__ bias, float* __restrict__ C,
    int M, int N, int Kt,
    int lda, int ldb, int ldc,
    long long batchA, long long batchB, long long batchC,
    int transB, int act, float alpha,
    int convMode, int Tconv, int Kin)
{
  __shared__ __align__(16) unsigned short sA[128 * 32];
  __shared__ __align__(16) unsigned short sB[64 * 32];   // transposed: [n][k]

  const int z = blockIdx.z;
  A  += (long long)z * batchA;
  Bw += (long long)z * batchB;
  C  += (long long)z * batchC;

  const int tid  = threadIdx.x;
  const int lane = tid & 31;
  const int wave = tid >> 5;
  const int wm   = wave & 3;     // 4 row groups of 32
  const int wn   = wave >> 2;    // 2 col groups of 32
  const int m0   = blockIdx.y * 128;
  const int n0   = blockIdx.x * 64;
  const int half = lane >> 4;
  const int l15  = lane & 15;

  v8f acc[2][2];
  #pragma unroll
  for (int i = 0; i < 2; ++i)
    #pragma unroll
    for (int j = 0; j < 2; ++j)
      #pragma unroll
      for (int r = 0; r < 8; ++r) acc[i][j][r] = 0.0f;

  for (int k0 = 0; k0 < Kt; k0 += 32) {
    // ---- stage A tile (128 x 32) ----
    #pragma unroll 4
    for (int e = tid; e < 128 * 32; e += 256) {
      int row = e >> 5, col = e & 31;
      int m = m0 + row; if (m >= M) m = M - 1;
      int k = k0 + col;
      float v = 0.0f;
      if (k < Kt) {
        if (convMode) {
          int bb = m / Tconv, t = m - bb * Tconv;
          int tap = k / Kin, ii = k - tap * Kin;
          int tt = t + tap - 1;
          if (tt >= 0 && tt < Tconv)
            v = A[(long long)(bb * Tconv + tt) * lda + ii];
        } else {
          v = A[(long long)m * lda + k];
        }
      }
      sA[e] = f2bf(v);
    }
    // ---- stage B tile transposed (64 cols x 32 k) ----
    #pragma unroll 4
    for (int e = tid; e < 64 * 32; e += 256) {
      int n = e >> 5, col = e & 31;
      int k = k0 + col;
      int gn = n0 + n;
      float v = 0.0f;
      if (k < Kt && gn < N) {
        if (convMode) {
          int tap = k / Kin, ii = k - tap * Kin;
          v = Bw[((long long)gn * Kin + ii) * 3 + tap];
        } else if (transB) {
          v = Bw[(long long)gn * ldb + k];
        } else {
          v = Bw[(long long)k * ldb + gn];
        }
      }
      sB[e] = f2bf(v);
    }
    __syncthreads();

    // ---- fragments per ISA VGPR layout, two 16B LDS loads each ----
    FragU fa[2], fb[2];
    #pragma unroll
    for (int i = 0; i < 2; ++i) {
      const u32x4* p = (const u32x4*)(sA + (wm * 32 + i * 16 + l15) * 32 + half * 8);
      fa[i].q[0] = p[0];          // K = kbase + 0..7
      fa[i].q[1] = p[2];          // K = kbase + 16..23
    }
    #pragma unroll
    for (int j = 0; j < 2; ++j) {
      const u32x4* p = (const u32x4*)(sB + (wn * 32 + j * 16 + l15) * 32 + half * 8);
      fb[j].q[0] = p[0];          // K = kbase + 0..7
      fb[j].q[1] = p[2];          // K = kbase + 16..23
    }
    #pragma unroll
    for (int i = 0; i < 2; ++i)
      #pragma unroll
      for (int j = 0; j < 2; ++j)
        acc[i][j] = __builtin_amdgcn_wmma_f32_16x16x32_bf16(
            false, fa[i].v, false, fb[j].v, (short)0, acc[i][j], false, false);
    __syncthreads();
  }

  // ---- epilogue: alpha, bias, relu, guarded store ----
  #pragma unroll
  for (int i = 0; i < 2; ++i) {
    #pragma unroll
    for (int j = 0; j < 2; ++j) {
      int nn = n0 + wn * 32 + j * 16 + l15;
      if (nn >= N) continue;
      float bval = bias ? bias[nn] : 0.0f;
      #pragma unroll
      for (int r = 0; r < 8; ++r) {
        int mm = m0 + wm * 32 + i * 16 + half * 8 + r;
        if (mm >= M) continue;
        float v = acc[i][j][r] * alpha + bval;
        if (act == 1) v = fmaxf(v, 0.0f);
        C[(long long)mm * ldc + nn] = v;
      }
    }
  }
}

// ---------------- elementwise / rowwise kernels ----------------
__global__ void embed_kernel(const int* __restrict__ inputs, const float* __restrict__ emb,
                             float* __restrict__ maskE, float* __restrict__ h) {
  int m = blockIdx.x;
  int t = m % S_;
  int tok = inputs[m];
  float mk = (tok != 0) ? 1.0f : 0.0f;   // PAD_IDX == 0
  if (threadIdx.x == 0) maskE[m] = mk;
  for (int d = threadIdx.x; d < D_; d += 128)
    h[(size_t)m * D_ + d] = emb[(size_t)tok * D_ + d] + pos_emb(t, d, D_) * mk;
}

__global__ void ln_kernel(const float* __restrict__ X, const float* __restrict__ R,
                          const float* __restrict__ g, const float* __restrict__ bb,
                          const float* __restrict__ mask, float* __restrict__ Out, int D) {
  size_t m = blockIdx.x;
  const float* x = X + m * (size_t)D;
  const float* r = R ? (R + m * (size_t)D) : nullptr;
  __shared__ float red[128];
  int tid = threadIdx.x;
  float loc[4];
  int cnt = 0;
  float s = 0.0f;
  for (int d = tid; d < D; d += 128) { float v = x[d] + (r ? r[d] : 0.0f); loc[cnt++] = v; s += v; }
  red[tid] = s; __syncthreads();
  for (int o = 64; o > 0; o >>= 1) { if (tid < o) red[tid] += red[tid + o]; __syncthreads(); }
  float mu = red[0] / (float)D; __syncthreads();
  float s2 = 0.0f;
  for (int c = 0; c < cnt; ++c) { float dd = loc[c] - mu; s2 += dd * dd; }
  red[tid] = s2; __syncthreads();
  for (int o = 64; o > 0; o >>= 1) { if (tid < o) red[tid] += red[tid + o]; __syncthreads(); }
  float inv = rsqrtf(red[0] / (float)D + 1e-5f);
  float mk = mask ? mask[m] : 1.0f;
  cnt = 0;
  for (int d = tid; d < D; d += 128)
    Out[m * (size_t)D + d] = ((loc[cnt++] - mu) * inv * g[d] + bb[d]) * mk;
}

__global__ void softmax_kernel(float* __restrict__ Sc, const float* __restrict__ keymask, int T) {
  int q = blockIdx.x, b = blockIdx.y;
  float* row = Sc + ((size_t)b * T + q) * T;
  const float* km = keymask + (size_t)b * T;
  __shared__ float red[128];
  int tid = threadIdx.x;
  float loc[4];
  int cnt = 0;
  float mx = -3.0e38f;
  for (int k = tid; k < T; k += 128) {
    float v = (km[k] > 0.0f) ? row[k] : -1.0e9f;
    loc[cnt++] = v;
    mx = fmaxf(mx, v);
  }
  red[tid] = mx; __syncthreads();
  for (int o = 64; o > 0; o >>= 1) { if (tid < o) red[tid] = fmaxf(red[tid], red[tid + o]); __syncthreads(); }
  mx = red[0]; __syncthreads();
  float s = 0.0f;
  for (int c = 0; c < cnt; ++c) { float e = expf(loc[c] - mx); loc[c] = e; s += e; }
  red[tid] = s; __syncthreads();
  for (int o = 64; o > 0; o >>= 1) { if (tid < o) red[tid] += red[tid + o]; __syncthreads(); }
  float inv = 1.0f / red[0];
  cnt = 0;
  for (int k = tid; k < T; k += 128) row[k] = loc[cnt++] * inv;
}

__global__ void dp_fc_kernel(const float* __restrict__ X, const float* __restrict__ w,
                             const float* __restrict__ b, const float* __restrict__ mask,
                             float* __restrict__ logdur, float* __restrict__ dur) {
  int wv = threadIdx.x >> 5, lane = threadIdx.x & 31;
  int m = blockIdx.x * 8 + wv;
  float s = 0.0f;
  for (int c = lane; c < DPF_; c += 32) s += X[(size_t)m * DPF_ + c] * w[c];
  for (int o = 16; o > 0; o >>= 1) s += __shfl_down(s, o, 32);
  if (lane == 0) {
    float ld = (s + b[0]) * mask[m];
    logdur[m] = ld;
    float d = expf(ld) - 1.0f;
    dur[m] = fminf(fmaxf(d, 0.0f), MAXDUR_);
  }
}

__global__ void lr_scan_kernel(const float* __restrict__ dur_tgt, int* __restrict__ cum,
                               int* __restrict__ declen) {
  int b = blockIdx.x, tid = threadIdx.x;   // blockDim = 256 = S_
  __shared__ int sc[S_];
  sc[tid] = (int)rintf(dur_tgt[b * S_ + tid] * (1.0f / PACE_));
  __syncthreads();
  for (int o = 1; o < S_; o <<= 1) {
    int v = (tid >= o) ? sc[tid - o] : 0;
    __syncthreads();
    sc[tid] += v;
    __syncthreads();
  }
  cum[b * S_ + tid] = sc[tid];
  if (tid == 0) { int tot = sc[S_ - 1]; declen[b] = tot < TDM_ ? tot : TDM_; }
}

__global__ void decmask_kernel(const int* __restrict__ declen, float* __restrict__ maskD,
                               float* __restrict__ outm) {
  int i = blockIdx.x * 256 + threadIdx.x;
  if (i >= B_ * TDM_) return;
  int b = i / TDM_, pos = i - b * TDM_;
  float v = (pos < declen[b]) ? 1.0f : 0.0f;
  maskD[i] = v;
  outm[i] = v;
}

__global__ void declens_out_kernel(const int* __restrict__ declen, float* __restrict__ o) {
  int i = threadIdx.x;
  if (i < B_) o[i] = (float)declen[i];
}

__global__ void build_y_kernel(const int* __restrict__ cum, const float* __restrict__ encout,
                               const float* __restrict__ maskD, float* __restrict__ h) {
  int m = blockIdx.x;                       // b*TDM_ + pos
  int b = m / TDM_, pos = m - b * TDM_;
  const int* cb = cum + b * S_;
  int lo = 0, hi = S_;
  while (lo < hi) { int mid = (lo + hi) >> 1; if (cb[mid] <= pos) lo = mid + 1; else hi = mid; }
  int idx = (lo < S_) ? lo : (S_ - 1);
  float mk = maskD[m];
  const float* src = encout + ((size_t)b * S_ + idx) * D_;
  for (int d = threadIdx.x; d < D_; d += 128) {
    float v = (mk > 0.0f) ? (src[d] + pos_emb(pos, d, D_)) : 40.0f;
    h[(size_t)m * D_ + d] = v;
  }
}

__global__ void copy_kernel(const float* __restrict__ src, float* __restrict__ dst, size_t n) {
  size_t i = (size_t)blockIdx.x * 256 + threadIdx.x;
  size_t stride = (size_t)gridDim.x * 256;
  for (; i < n; i += stride) dst[i] = src[i];
}

// ---------------- host orchestration ----------------
extern "C" void kernel_launch(void* const* d_in, const int* in_sizes, int n_in,
                              void* d_out, int out_size, void* d_ws, size_t ws_size,
                              hipStream_t stream) {
  (void)in_sizes; (void)n_in; (void)out_size; (void)ws_size;

  const int*   inputs  = (const int*)  d_in[0];
  const float* dur_tgt = (const float*)d_in[1];
  const float* emb     = (const float*)d_in[2];
  const float* enc_w[11]; for (int i = 0; i < 11; ++i) enc_w[i] = (const float*)d_in[3 + i];
  const float* dec_w[11]; for (int i = 0; i < 11; ++i) dec_w[i] = (const float*)d_in[14 + i];
  const float* dp_c1_w = (const float*)d_in[25];
  const float* dp_c1_b = (const float*)d_in[26];
  const float* dp_ln1g = (const float*)d_in[27];
  const float* dp_ln1b = (const float*)d_in[28];
  const float* dp_c2_w = (const float*)d_in[29];
  const float* dp_c2_b = (const float*)d_in[30];
  const float* dp_ln2g = (const float*)d_in[31];
  const float* dp_ln2b = (const float*)d_in[32];
  const float* dp_fc_w = (const float*)d_in[33];
  const float* dp_fc_b = (const float*)d_in[34];
  const float* proj_w  = (const float*)d_in[35];
  const float* proj_b  = (const float*)d_in[36];

  const int ME = B_ * S_;     // 8192
  const int MD = B_ * TDM_;   // 12800

  float* out         = (float*)d_out;
  float* mel_out     = out;                               // 32*400*80
  float* out_decmask = mel_out + (size_t)B_ * TDM_ * NMEL_;
  float* out_dur     = out_decmask + (size_t)B_ * TDM_;
  float* out_logdur  = out_dur + (size_t)B_ * S_;
  float* out_declens = out_logdur + (size_t)B_ * S_;

  float* ws = (float*)d_ws;
  size_t off = 0;
  float* h      = ws + off; off += (size_t)MD * D_;
  float* tmp    = ws + off; off += (size_t)MD * D_;
  float* encout = ws + off; off += (size_t)ME * D_;
  float* bigR   = ws + off; off += (size_t)MD * DI_;      // union region
  float* maskE  = ws + off; off += ME;
  float* maskD  = ws + off; off += MD;
  int*   cum    = (int*)(ws + off); off += ME;
  int*   declen = (int*)(ws + off); off += 64;

  float* qkv    = bigR;
  float* scores = qkv + (size_t)MD * A3_;
  float* vec    = scores + (size_t)B_ * TDM_ * TDM_;
  float* cbuf   = bigR;                                   // conv phase
  float* dpa    = bigR;                                   // DP phase
  float* dpb    = bigR + (size_t)ME * DPF_;

  auto gemm = [&](const float* A, const float* Bw, const float* bias, float* C,
                  int M, int N, int Kt, int lda, int ldb, int ldc,
                  long long bA, long long bB, long long bC, int batches,
                  int transB, int act, float alpha, int convMode, int Tc, int Kin) {
    dim3 g((N + 63) / 64, (M + 127) / 128, batches);
    wmma_gemm<<<g, dim3(256), 0, stream>>>(A, Bw, bias, C, M, N, Kt, lda, ldb, ldc,
                                           bA, bB, bC, transB, act, alpha, convMode, Tc, Kin);
  };

  auto fft_layer = [&](float* hbuf, const float* mask, int T, int M, const float* const* W, int l) {
    const float* qkvw = W[0] + (size_t)l * D_ * A3_;
    const float* qkvb = W[1] + (size_t)l * A3_;
    const float* ow   = W[2] + (size_t)l * DH_ * D_;
    const float* l1g  = W[3] + (size_t)l * D_;
    const float* l1b  = W[4] + (size_t)l * D_;
    const float* c1w  = W[5] + (size_t)l * DI_ * D_ * KW_;
    const float* c1b  = W[6] + (size_t)l * DI_;
    const float* c2w  = W[7] + (size_t)l * D_ * DI_ * KW_;
    const float* c2b  = W[8] + (size_t)l * D_;
    const float* l2g  = W[9] + (size_t)l * D_;
    const float* l2b  = W[10] + (size_t)l * D_;

    // qkv = h @ qkv_w + qkv_b
    gemm(hbuf, qkvw, qkvb, qkv, M, A3_, D_, D_, A3_, A3_, 0, 0, 0, 1, 0, 0, 1.0f, 0, 0, 1);
    // scores = (q @ k^T) * scale  (batched over B)
    gemm(qkv, qkv + DH_, nullptr, scores, T, T, DH_, A3_, A3_, T,
         (long long)T * A3_, (long long)T * A3_, (long long)T * T, B_, 1, 0, ATTNSC_, 0, 0, 1);
    softmax_kernel<<<dim3(T, B_), dim3(128), 0, stream>>>(scores, mask, T);
    // vec = probs @ v
    gemm(scores, qkv + 2 * DH_, nullptr, vec, T, DH_, T, T, A3_, DH_,
         (long long)T * T, (long long)T * A3_, (long long)T * DH_, B_, 0, 0, 1.0f, 0, 0, 1);
    // attn-out = vec @ o_w
    gemm(vec, ow, nullptr, tmp, M, D_, DH_, DH_, D_, D_, 0, 0, 0, 1, 0, 0, 1.0f, 0, 0, 1);
    ln_kernel<<<dim3(M), dim3(128), 0, stream>>>(hbuf, tmp, l1g, l1b, mask, hbuf, D_);
    // conv FFN (im2col fused)
    gemm(hbuf, c1w, c1b, cbuf, M, DI_, KW_ * D_, D_, 0, DI_, 0, 0, 0, 1, 0, 1, 1.0f, 1, T, D_);
    gemm(cbuf, c2w, c2b, tmp, M, D_, KW_ * DI_, DI_, 0, D_, 0, 0, 0, 1, 0, 0, 1.0f, 1, T, DI_);
    ln_kernel<<<dim3(M), dim3(128), 0, stream>>>(hbuf, tmp, l2g, l2b, mask, hbuf, D_);
  };

  // ---- encoder ----
  embed_kernel<<<dim3(ME), dim3(128), 0, stream>>>(inputs, emb, maskE, h);
  for (int l = 0; l < NL_; ++l) fft_layer(h, maskE, S_, ME, enc_w, l);
  copy_kernel<<<dim3(4096), dim3(256), 0, stream>>>(h, encout, (size_t)ME * D_);

  // ---- duration predictor ----
  gemm(encout, dp_c1_w, dp_c1_b, dpa, ME, DPF_, KW_ * D_, D_, 0, DPF_, 0, 0, 0, 1, 0, 1, 1.0f, 1, S_, D_);
  ln_kernel<<<dim3(ME), dim3(128), 0, stream>>>(dpa, nullptr, dp_ln1g, dp_ln1b, nullptr, dpb, DPF_);
  gemm(dpb, dp_c2_w, dp_c2_b, dpa, ME, DPF_, KW_ * DPF_, DPF_, 0, DPF_, 0, 0, 0, 1, 0, 1, 1.0f, 1, S_, DPF_);
  ln_kernel<<<dim3(ME), dim3(128), 0, stream>>>(dpa, nullptr, dp_ln2g, dp_ln2b, nullptr, dpb, DPF_);
  dp_fc_kernel<<<dim3(ME / 8), dim3(256), 0, stream>>>(dpb, dp_fc_w, dp_fc_b, maskE, out_logdur, out_dur);

  // ---- length regulator ----
  lr_scan_kernel<<<dim3(B_), dim3(S_), 0, stream>>>(dur_tgt, cum, declen);
  decmask_kernel<<<dim3((B_ * TDM_ + 255) / 256), dim3(256), 0, stream>>>(declen, maskD, out_decmask);
  declens_out_kernel<<<dim3(1), dim3(64), 0, stream>>>(declen, out_declens);
  build_y_kernel<<<dim3(MD), dim3(128), 0, stream>>>(cum, encout, maskD, h);

  // ---- decoder ----
  for (int l = 0; l < NL_; ++l) fft_layer(h, maskD, TDM_, MD, dec_w, l);

  // ---- mel projection ----
  gemm(h, proj_w, proj_b, mel_out, MD, NMEL_, D_, D_, NMEL_, NMEL_, 0, 0, 0, 1, 0, 0, 1.0f, 0, 0, 1);
}
